// MSIB_57724360458772
// MI455X (gfx1250) — matrix operand order repaired and checked by reading, not compiled, not run on hardware
//
#include <hip/hip_runtime.h>

// ---------------------------------------------------------------------------
// MSIB forward for MI455X (gfx1250, wave32).
//
// Dense MLPs -> v_wmma_f32_16x16x32_f16 (f16 multiplicands, f32 accumulate),
// with weights pre-swizzled into WMMA fragment-major order (one 32B
// contiguous chunk per lane per fragment -> b128 loads) and A fragments
// hoisted out of the column-tile loop (loaded once per wave as float4s).
// GEMM shapes are templated so the K loop fully unrolls into back-to-back
// WMMAs. Everything else (GIN aggregation, segment max/sum, hypergraph
// incidence scatter/gather) is memory-bound float-atomic traffic.
//
// Input index assumption: top-level setup_inputs() dict in insertion order
// (x, params, edge_index, batch, hyperedge_index, edge_batch, num_graphs,
// num_hyperedges); 'params' flattened jax-tree style (dict keys sorted,
// lists in order). Integer arrays are int32 (jax x32 default).
// ---------------------------------------------------------------------------

#define NN      50000
#define NE      400000
#define NG      128
#define FIN     32
#define HID     64
#define EMB     192
#define NH_MAX  (NN + NE)     // num_hyperedges = max_edge + E + 1 <= N + E
#define EPS_F   1e-10f
#define INV_2S  (1.0f / 40.0f) // 1/(2*SCALAR)

typedef __attribute__((ext_vector_type(16))) _Float16 v16h;
typedef __attribute__((ext_vector_type(8)))  float    v8f;

enum {
  IN_X = 0,
  IN_ENC0_B1, IN_ENC0_B2, IN_ENC0_W1, IN_ENC0_W2,
  IN_ENC1_B1, IN_ENC1_B2, IN_ENC1_W1, IN_ENC1_W2,
  IN_ENC2_B1, IN_ENC2_B2, IN_ENC2_W1, IN_ENC2_W2,
  IN_EXT0_B1, IN_EXT0_B2, IN_EXT0_W1, IN_EXT0_W2,
  IN_EXT1_B1, IN_EXT1_B2, IN_EXT1_W1, IN_EXT1_W2,
  IN_EXTMLP_B, IN_EXTMLP_W,
  IN_HYP0_B, IN_HYP0_W, IN_HYP1_B, IN_HYP1_W, IN_HYP2_B, IN_HYP2_W,
  IN_PROJ_B1, IN_PROJ_B2, IN_PROJ_W1, IN_PROJ_W2,
  IN_PROJH_B1, IN_PROJH_B2, IN_PROJH_W1, IN_PROJH_W2,
  IN_EDGE_INDEX, IN_BATCH, IN_HYPEREDGE_INDEX, IN_EDGE_BATCH,
  IN_NUM_GRAPHS, IN_NUM_HYPEREDGES
};

// ---------------------------------------------------------------------------
// Weight pre-swizzle: row-major f32 W[fin,fout] -> f16 fragment-major layout.
// Fragment (kc, nt) holds the 32x16 B tile for K chunk kc, column tile nt, in
// the exact wave32 VGPR layout of CDNA5 ISA §7.12.2:
//   lane l: n = l&15, half = l>>4; element i (0..15) = W[kc*32 + 8*half +
//   (i&7) + 16*(i>>3), nt*16 + n].  Stored as frag[(kc*ntiles + nt)*32 + lane][i] (32B/lane).
// ---------------------------------------------------------------------------
__global__ void k_pack_w(const float* __restrict__ w, _Float16* __restrict__ d,
                         int fin, int fout)
{
  int t = blockIdx.x * blockDim.x + threadIdx.x;
  if (t >= fin * fout) return;
  int i    = t & 15;
  int lane = (t >> 4) & 31;
  int ft   = t >> 9;                 // fragment index = kc*ntiles + nt
  int ntiles = fout >> 4;
  int kc = ft / ntiles, nt = ft - kc * ntiles;
  int n = lane & 15, half = lane >> 4;
  int k = kc * 32 + 8 * half + (i & 7) + 16 * (i >> 3);
  d[t] = (_Float16)w[(size_t)k * fout + nt * 16 + n];
}

// ---------------------------------------------------------------------------
// WMMA GEMM: Out[nrows,TFOUT] = act(In[nrows,TFIN] @ W + bias)
// One wave computes 16 rows. A fragments (K chunks) loaded once per wave via
// float4 (b128) loads + f16 convert; B fragments are single 32B contiguous
// loads from the pre-swizzled weight arena. nrows % 16 == 0 (N, E, G all are).
// ---------------------------------------------------------------------------
template <int TFIN, int TFOUT>
__global__ void k_wmma_gemm(const float* __restrict__ In, int ldin,
                            const _Float16* __restrict__ Wf,
                            const float* __restrict__ bias,
                            float* __restrict__ Out, int ldout,
                            int nrows, int relu_act)
{
  constexpr int KC = TFIN / 32;      // K chunks
  constexpr int NT = TFOUT / 16;     // column tiles
  const int lane = threadIdx.x & 31;
  const int wave = blockIdx.x * (blockDim.x >> 5) + (threadIdx.x >> 5);
  const int row0 = wave << 4;
  if (row0 >= nrows) return;         // uniform per wave: EXEC stays all-1s
  const int m    = lane & 15;
  const int half = lane >> 4;

  // --- load all A fragments once (independent of column tile) ---
  v16h afrag[KC];
  const float* rowp = In + (size_t)(row0 + m) * ldin + 8 * half;
#pragma unroll
  for (int kc = 0; kc < KC; ++kc) {
    const float4* p0 = (const float4*)(rowp + kc * 32);       // K = 8h+0..7
    const float4* p1 = (const float4*)(rowp + kc * 32 + 16);  // K = 16+8h+0..7
    float4 x0 = p0[0], x1 = p0[1];
    float4 y0 = p1[0], y1 = p1[1];
    v16h a;
    a[0]  = (_Float16)x0.x; a[1]  = (_Float16)x0.y;
    a[2]  = (_Float16)x0.z; a[3]  = (_Float16)x0.w;
    a[4]  = (_Float16)x1.x; a[5]  = (_Float16)x1.y;
    a[6]  = (_Float16)x1.z; a[7]  = (_Float16)x1.w;
    a[8]  = (_Float16)y0.x; a[9]  = (_Float16)y0.y;
    a[10] = (_Float16)y0.z; a[11] = (_Float16)y0.w;
    a[12] = (_Float16)y1.x; a[13] = (_Float16)y1.y;
    a[14] = (_Float16)y1.z; a[15] = (_Float16)y1.w;
    afrag[kc] = a;
  }

  const v16h* wfrag = (const v16h*)Wf;
  for (int nt = 0; nt < NT; ++nt) {
    v8f c = {};
#pragma unroll
    for (int kc = 0; kc < KC; ++kc) {
      v16h b = wfrag[(size_t)(kc * NT + nt) * 32 + lane];
      c = __builtin_amdgcn_wmma_f32_16x16x32_f16(
              /*neg_a=*/false, afrag[kc], /*neg_b=*/false, b,
              /*c_mod=*/(short)0, c, /*reuse_a=*/false, /*reuse_b=*/false);
    }
    const float bb = bias ? bias[nt * 16 + m] : 0.0f;
#pragma unroll
    for (int r = 0; r < 8; ++r) {
      float v = c[r] + bb;
      if (relu_act) v = fmaxf(v, 0.0f);
      Out[(size_t)(row0 + 8 * half + r) * ldout + nt * 16 + m] = v;
    }
  }
}

// --------------------------- elementwise helpers ---------------------------
__global__ void k_copy_cols(const float* __restrict__ in, int ldin, int coloff,
                            float* __restrict__ out, int ldout, long long total, int cols)
{
  long long t = (long long)blockIdx.x * blockDim.x + threadIdx.x;
  if (t >= total) return;
  int i = (int)(t / cols), c = (int)(t % cols);
  out[(size_t)i * ldout + c] = in[(size_t)i * ldin + coloff + c];
}

// GIN aggregation: z[dst] += h[src] (z pre-initialized with h itself)
__global__ void k_edge_scatter(const float* __restrict__ h, int ldh, int coloff,
                               const int* __restrict__ src, const int* __restrict__ dst,
                               float* __restrict__ z, int ldz, long long total, int cols)
{
  long long t = (long long)blockIdx.x * blockDim.x + threadIdx.x;
  if (t >= total) return;
  int e = (int)(t / cols), c = (int)(t % cols);
  atomicAdd(&z[(size_t)dst[e] * ldz + c], h[(size_t)src[e] * ldh + coloff + c]);
}

__global__ void k_node_imp(const float* __restrict__ xs, const float* __restrict__ w,
                           const float* __restrict__ b, float* __restrict__ ni, int n)
{
  int i = blockIdx.x * blockDim.x + threadIdx.x;
  if (i >= n) return;
  float acc = b[0];
  const float* r = xs + (size_t)i * 128;
#pragma unroll 8
  for (int c = 0; c < 128; ++c) acc += r[c] * w[c];
  ni[i] = 1.0f / (1.0f + __expf(-acc));
}

// segment max of strictly-positive floats: uint bit pattern is monotonic
__global__ void k_segmax(const float* __restrict__ v, const int* __restrict__ seg,
                         float* __restrict__ out, int n)
{
  int i = blockIdx.x * blockDim.x + threadIdx.x;
  if (i < n) atomicMax(reinterpret_cast<unsigned int*>(out + seg[i]), __float_as_uint(v[i]));
}

__global__ void k_scale_x(const float* __restrict__ x, const float* __restrict__ ni,
                          const float* __restrict__ gm, const int* __restrict__ batch,
                          float* __restrict__ out, long long total)
{
  long long t = (long long)blockIdx.x * blockDim.x + threadIdx.x;
  if (t >= total) return;
  int i = (int)(t / FIN);
  float s = ni[i] / (gm[batch[i]] + EPS_F);
  s = (2.0f * s - 1.0f) * INV_2S + 1.0f;
  out[t] = x[t] * s;
}

__global__ void k_graph_sum(const float* __restrict__ v, int ld, const int* __restrict__ seg,
                            float* __restrict__ out, long long total, int cols)
{
  long long t = (long long)blockIdx.x * blockDim.x + threadIdx.x;
  if (t >= total) return;
  int i = (int)(t / cols), c = (int)(t % cols);
  atomicAdd(&out[(size_t)seg[i] * cols + c], v[(size_t)i * ld + c]);
}

__global__ void k_edge_imp(const float* __restrict__ ni, const int* __restrict__ src,
                           const int* __restrict__ dst, const int* __restrict__ eb,
                           float* __restrict__ eimp, float* __restrict__ gme, int nE)
{
  int e = blockIdx.x * blockDim.x + threadIdx.x;
  if (e >= nE) return;
  float v = ni[src[e]] * ni[dst[e]];
  eimp[e] = v;
  atomicMax(reinterpret_cast<unsigned int*>(gme + eb[e]), __float_as_uint(v));
}

__global__ void k_edge_attr(const float* __restrict__ x, const int* __restrict__ src,
                            const int* __restrict__ dst, const float* __restrict__ eimp,
                            const float* __restrict__ gme, const int* __restrict__ eb,
                            float* __restrict__ ea, long long total)
{
  long long t = (long long)blockIdx.x * blockDim.x + threadIdx.x;
  if (t >= total) return;
  int e = (int)(t / FIN), c = (int)(t % FIN);
  float se = eimp[e] / (gme[eb[e]] + EPS_F);
  se = (2.0f * se - 1.0f) * INV_2S + 1.0f;
  ea[(size_t)e * HID + c] =
      0.5f * (x[(size_t)src[e] * FIN + c] + x[(size_t)dst[e] * FIN + c]) * se;
}

__global__ void k_degree(const int* __restrict__ idx, float* __restrict__ deg, int nnz)
{
  int j = blockIdx.x * blockDim.x + threadIdx.x;
  if (j < nnz) atomicAdd(&deg[idx[j]], 1.0f);
}

__global__ void k_invert(float* __restrict__ d, int n)
{
  int i = blockIdx.x * blockDim.x + threadIdx.x;
  if (i < n) { float v = d[i]; d[i] = (v > 0.0f) ? 1.0f / v : 0.0f; }
}

__global__ void k_hyper_scatter(const float* __restrict__ xw, const int* __restrict__ ni,
                                const int* __restrict__ ei, float* __restrict__ he,
                                long long total)
{
  long long t = (long long)blockIdx.x * blockDim.x + threadIdx.x;
  if (t >= total) return;
  int j = (int)(t >> 6), c = (int)(t & 63);
  atomicAdd(&he[(size_t)ei[j] * HID + c], xw[(size_t)ni[j] * HID + c]);
}

// Binv folded into the gather read (saves a full HE rescale pass)
__global__ void k_hyper_gather(const float* __restrict__ he, const float* __restrict__ binv,
                               const int* __restrict__ ni, const int* __restrict__ ei,
                               float* __restrict__ out, long long total)
{
  long long t = (long long)blockIdx.x * blockDim.x + threadIdx.x;
  if (t >= total) return;
  int j = (int)(t >> 6), c = (int)(t & 63);
  int h = ei[j];
  atomicAdd(&out[(size_t)ni[j] * HID + c], he[(size_t)h * HID + c] * binv[h]);
}

// h = relu(acc*Dinv + bias); write both next-layer buffer and y_hyper_node slice
__global__ void k_hyper_finalize(float* __restrict__ acc, const float* __restrict__ dinv,
                                 const float* __restrict__ bias, float* __restrict__ ynode,
                                 int coloff, long long total)
{
  long long t = (long long)blockIdx.x * blockDim.x + threadIdx.x;
  if (t >= total) return;
  int e = (int)(t >> 6), c = (int)(t & 63);
  float v = fmaxf(acc[t] * dinv[e] + bias[c], 0.0f);
  acc[t] = v;
  ynode[(size_t)e * EMB + coloff + c] = v;
}

// ---------------------------------------------------------------------------
static inline unsigned cdivu(long long a, int b) { return (unsigned)((a + b - 1) / b); }
static inline size_t   alignup(size_t x)         { return (x + 255) & ~(size_t)255; }

extern "C" void kernel_launch(void* const* d_in, const int* in_sizes, int n_in,
                              void* d_out, int out_size, void* d_ws, size_t ws_size,
                              hipStream_t stream)
{
  (void)n_in; (void)out_size; (void)ws_size;

  const float* x      = (const float*)d_in[IN_X];
  const int*   eidx   = (const int*)d_in[IN_EDGE_INDEX];
  const int*   esrc   = eidx;
  const int*   edst   = eidx + NE;
  const int*   batch  = (const int*)d_in[IN_BATCH];
  const int*   hidx   = (const int*)d_in[IN_HYPEREDGE_INDEX];
  const int    nnz    = in_sizes[IN_HYPEREDGE_INDEX] / 2;
  const int*   hni    = hidx;          // hypergraph node ids (original edges)
  const int*   hhe    = hidx + nnz;    // hyperedge ids
  const int*   ebatch = (const int*)d_in[IN_EDGE_BATCH];

  // ---------------- workspace layout ----------------
  char* ws = (char*)d_ws; size_t off = 0;
  auto alloc = [&](size_t bytes) -> char* { char* p = ws + off; off = alignup(off + bytes); return p; };
  float* ZB  = (float*)alloc((size_t)NN * HID * 4);       // GIN pre-MLP sums
  float* TB  = (float*)alloc((size_t)NN * HID * 4);       // MLP hidden
  float* XS  = (float*)alloc((size_t)NN * 128 * 4);       // extractor concat
  float* NI  = (float*)alloc((size_t)NN * 4);             // node importance
  float* GM  = (float*)alloc((size_t)NG * 4);             // per-graph node max
  float* GME = (float*)alloc((size_t)NG * 4);             // per-graph edge max
  float* XSC = (float*)alloc((size_t)NN * FIN * 4);       // scaled x
  float* EIM = (float*)alloc((size_t)NE * 4);             // edge importance
  float* EA  = (float*)alloc((size_t)NE * HID * 4);       // hyper h (in/out)
  float* EZ  = (float*)alloc((size_t)NE * HID * 4);       // hyper x@W
  float* HE  = (float*)alloc((size_t)NH_MAX * HID * 4);   // hyperedge sums
  float* DV  = (float*)alloc((size_t)NE * 4);             // Dinv
  float* BV  = (float*)alloc((size_t)NH_MAX * 4);         // Binv
  float* Y   = (float*)alloc((size_t)NG * EMB * 4);
  float* YHG = (float*)alloc((size_t)NG * EMB * 4);
  float* TG  = (float*)alloc((size_t)NG * EMB * 4);

  // f16 fragment-swizzled weight copies (rebuilt every call; deterministic)
  const int widx[17] = {IN_EXT0_W1, IN_EXT0_W2, IN_EXT1_W1, IN_EXT1_W2,
                        IN_ENC0_W1, IN_ENC0_W2, IN_ENC1_W1, IN_ENC1_W2,
                        IN_ENC2_W1, IN_ENC2_W2,
                        IN_HYP0_W, IN_HYP1_W, IN_HYP2_W,
                        IN_PROJ_W1, IN_PROJ_W2, IN_PROJH_W1, IN_PROJH_W2};
  const int wfin[17] = {32, 64, 64, 64, 32, 64, 64, 64, 64, 64, 32, 64, 64,
                        192, 192, 192, 192};
  const int wfout[17]= {64, 64, 64, 64, 64, 64, 64, 64, 64, 64, 64, 64, 64,
                        192, 192, 192, 192};
  _Float16* WH[17];
  for (int i = 0; i < 17; ++i) {
    int n = wfin[i] * wfout[i];
    WH[i] = (_Float16*)alloc((size_t)n * 2);
    k_pack_w<<<cdivu(n, 256), 256, 0, stream>>>((const float*)d_in[widx[i]], WH[i],
                                                wfin[i], wfout[i]);
  }

  // ---------------- output layout ----------------
  float* OUT_RY  = (float*)d_out;                         // raw_y       [G,192]
  float* OUT_RYH = OUT_RY + (size_t)NG * EMB;             // raw_y_hyper [G,192]
  float* YN      = OUT_RYH + (size_t)NG * EMB;            // y_node      [N,192]
  float* YHN     = YN + (size_t)NN * EMB;                 // y_hyper_node[E,192]

  auto gemm = [&](const float* In, int ldin, const _Float16* Wf, const float* bias,
                  float* Out, int ldout, int nrows, int fin, int fout, int act) {
    unsigned blocks = (unsigned)(nrows / 16 + 7) / 8;     // 8 waves per block
    if (fin == 32 && fout == 64)
      k_wmma_gemm<32, 64><<<blocks, 256, 0, stream>>>(In, ldin, Wf, bias, Out, ldout, nrows, act);
    else if (fin == 64 && fout == 64)
      k_wmma_gemm<64, 64><<<blocks, 256, 0, stream>>>(In, ldin, Wf, bias, Out, ldout, nrows, act);
    else
      k_wmma_gemm<192, 192><<<blocks, 256, 0, stream>>>(In, ldin, Wf, bias, Out, ldout, nrows, act);
  };

  const long long NnH = (long long)NN * HID, NeH = (long long)NE * HID;
  const long long NnF = (long long)NN * FIN, NeF = (long long)NE * FIN;
  const long long NzH = (long long)nnz * HID;

  // ================= extractor GIN =================
  // layer 0: z = x + agg(x); h0 = relu(relu(z@w1+b1)@w2+b2) -> XS[:,0:64]
  k_copy_cols<<<cdivu(NnF, 256), 256, 0, stream>>>(x, FIN, 0, ZB, HID, NnF, FIN);
  k_edge_scatter<<<cdivu(NeF, 256), 256, 0, stream>>>(x, FIN, 0, esrc, edst, ZB, HID, NeF, FIN);
  gemm(ZB, HID, WH[0], (const float*)d_in[IN_EXT0_B1], TB, HID, NN, FIN, HID, 1);
  gemm(TB, HID, WH[1], (const float*)d_in[IN_EXT0_B2], XS, 128, NN, HID, HID, 1);
  // layer 1: z = h0 + agg(h0); h1 = relu(z@w1+b1)@w2+b2 (no outer relu) -> XS[:,64:128]
  k_copy_cols<<<cdivu(NnH, 256), 256, 0, stream>>>(XS, 128, 0, ZB, HID, NnH, HID);
  k_edge_scatter<<<cdivu(NeH, 256), 256, 0, stream>>>(XS, 128, 0, esrc, edst, ZB, HID, NeH, HID);
  gemm(ZB, HID, WH[2], (const float*)d_in[IN_EXT1_B1], TB, HID, NN, HID, HID, 1);
  gemm(TB, HID, WH[3], (const float*)d_in[IN_EXT1_B2], XS + 64, 128, NN, HID, HID, 0);
  // node importance
  k_node_imp<<<cdivu(NN, 256), 256, 0, stream>>>(XS, (const float*)d_in[IN_EXTMLP_W],
                                                 (const float*)d_in[IN_EXTMLP_B], NI, NN);

  // ================= node rescale =================
  hipMemsetAsync(GM, 0, (size_t)NG * 4, stream);
  hipMemsetAsync(GME, 0, (size_t)NG * 4, stream);
  k_segmax<<<cdivu(NN, 256), 256, 0, stream>>>(NI, batch, GM, NN);
  k_scale_x<<<cdivu(NnF, 256), 256, 0, stream>>>(x, NI, GM, batch, XSC, NnF);

  // ================= encoder GIN (3 layers) =================
  for (int l = 0; l < 3; ++l) {
    const float* hsrc = (l == 0) ? XSC : YN;
    int ldh  = (l == 0) ? FIN : EMB;
    int coff = (l == 0) ? 0 : 64 * (l - 1);
    int cols = (l == 0) ? FIN : HID;
    long long tc = (long long)NN * cols, te = (long long)NE * cols;
    k_copy_cols<<<cdivu(tc, 256), 256, 0, stream>>>(hsrc, ldh, coff, ZB, HID, tc, cols);
    k_edge_scatter<<<cdivu(te, 256), 256, 0, stream>>>(hsrc, ldh, coff, esrc, edst, ZB, HID, te, cols);
    const _Float16* w1 = WH[4 + 2 * l];
    const _Float16* w2 = WH[5 + 2 * l];
    const float* b1 = (const float*)d_in[(l == 0) ? IN_ENC0_B1 : (l == 1) ? IN_ENC1_B1 : IN_ENC2_B1];
    const float* b2 = (const float*)d_in[(l == 0) ? IN_ENC0_B2 : (l == 1) ? IN_ENC1_B2 : IN_ENC2_B2];
    gemm(ZB, HID, w1, b1, TB, HID, NN, cols, HID, 1);
    gemm(TB, HID, w2, b2, YN + 64 * l, EMB, NN, HID, HID, 1);  // outer relu
  }
  hipMemsetAsync(Y, 0, (size_t)NG * EMB * 4, stream);
  k_graph_sum<<<cdivu((long long)NN * EMB, 256), 256, 0, stream>>>(YN, EMB, batch, Y,
                                                                   (long long)NN * EMB, EMB);

  // ================= hypergraph branch =================
  k_edge_imp<<<cdivu(NE, 256), 256, 0, stream>>>(NI, esrc, edst, ebatch, EIM, GME, NE);
  k_edge_attr<<<cdivu(NeF, 256), 256, 0, stream>>>(x, esrc, edst, EIM, GME, ebatch, EA, NeF);
  hipMemsetAsync(DV, 0, (size_t)NE * 4, stream);
  hipMemsetAsync(BV, 0, (size_t)NH_MAX * 4, stream);
  k_degree<<<cdivu(nnz, 256), 256, 0, stream>>>(hni, DV, nnz);
  k_degree<<<cdivu(nnz, 256), 256, 0, stream>>>(hhe, BV, nnz);
  k_invert<<<cdivu(NE, 256), 256, 0, stream>>>(DV, NE);
  k_invert<<<cdivu(NH_MAX, 256), 256, 0, stream>>>(BV, NH_MAX);

  for (int l = 0; l < 3; ++l) {
    int fin = (l == 0) ? FIN : HID;
    const _Float16* w = WH[10 + l];
    const float* b = (const float*)d_in[(l == 0) ? IN_HYP0_B : (l == 1) ? IN_HYP1_B : IN_HYP2_B];
    gemm(EA, HID, w, nullptr, EZ, HID, NE, fin, HID, 0);      // xw = h @ W
    hipMemsetAsync(HE, 0, (size_t)NH_MAX * HID * 4, stream);
    k_hyper_scatter<<<cdivu(NzH, 256), 256, 0, stream>>>(EZ, hni, hhe, HE, NzH);
    hipMemsetAsync(EA, 0, (size_t)NE * HID * 4, stream);
    k_hyper_gather<<<cdivu(NzH, 256), 256, 0, stream>>>(HE, BV, hni, hhe, EA, NzH);
    k_hyper_finalize<<<cdivu(NeH, 256), 256, 0, stream>>>(EA, DV, b, YHN, 64 * l, NeH);
  }
  hipMemsetAsync(YHG, 0, (size_t)NG * EMB * 4, stream);
  k_graph_sum<<<cdivu((long long)NE * EMB, 256), 256, 0, stream>>>(YHN, EMB, ebatch, YHG,
                                                                   (long long)NE * EMB, EMB);

  // ================= projections =================
  gemm(Y, EMB, WH[13], (const float*)d_in[IN_PROJ_B1], TG, EMB, NG, EMB, EMB, 1);
  gemm(TG, EMB, WH[14], (const float*)d_in[IN_PROJ_B2], OUT_RY, EMB, NG, EMB, EMB, 0);
  gemm(YHG, EMB, WH[15], (const float*)d_in[IN_PROJH_B1], TG, EMB, NG, EMB, EMB, 1);
  gemm(TG, EMB, WH[16], (const float*)d_in[IN_PROJH_B2], OUT_RYH, EMB, NG, EMB, EMB, 0);
}
